// AntiAttentionSegment_25829933318619
// MI455X (gfx1250) — matrix operand order repaired and checked by reading, not compiled, not run on hardware
//
#include <hip/hip_runtime.h>
#include <hip/hip_bf16.h>
#include <math.h>

// ---------------------------------------------------------------------------
// Problem constants (from reference): B=8, SEG=16, P=256, H=1024, NH=16
// ---------------------------------------------------------------------------
#define kB   8
#define kSEG 16
#define kH   1024
#define kNH  16
#define kDH  64
#define kT   257          // tokens per segment
#define kL   4113         // 1 + SEG*T
#define kTP  288          // padded T (9 * 32) for WMMA K-steps in P@V
#define LN_EPS 1e-12f

typedef __bf16 bf16;
typedef __attribute__((ext_vector_type(16))) __bf16 v16bf;
typedef __attribute__((ext_vector_type(8)))  __bf16 v8bf;
typedef __attribute__((ext_vector_type(8)))  float  v8f;

union V16U { v16bf v; v8bf h[2]; };

static __device__ __forceinline__ v8f wmma_bf16(const V16U& a, const V16U& b, v8f c) {
  // D = A(16x32 bf16) x B(32x16 bf16) + C(16x16 f32)
  return __builtin_amdgcn_wmma_f32_16x16x32_bf16(
      false, a.v, false, b.v, (short)0, c, false, false);
}

// ---------------------------------------------------------------------------
// Elementwise helpers
// ---------------------------------------------------------------------------
__global__ void k_f32_to_bf16(const float* __restrict__ in, bf16* __restrict__ out, long n) {
  long i = (long)blockIdx.x * blockDim.x + threadIdx.x;
  long st = (long)gridDim.x * blockDim.x;
  for (; i < n; i += st) out[i] = (bf16)in[i];
}

__global__ void k_fill_bf16_zero(bf16* __restrict__ p, long n) {
  long i = (long)blockIdx.x * blockDim.x + threadIdx.x;
  long st = (long)gridDim.x * blockDim.x;
  for (; i < n; i += st) p[i] = (bf16)0.0f;
}

// ---------------------------------------------------------------------------
// Generic bf16 GEMM:  C[m,n] = sum_k A[m,k] * W[n,k]   (x @ W.T), K = N = 1024
// Block tile 128x128, K-step 32, 256 threads = 8 waves, each wave: 4x2 WMMA
// accumulators (64x32 region).  Register double-buffered global->LDS staging:
// the next K-slice's global_load_b128s are issued before consuming the
// current LDS tile, so their latency hides under the 64 WMMAs per step.
// ---------------------------------------------------------------------------
struct AMapSegX {           // A rows = hidden_states[b, 1 + z*T + t, :], m = b*T + t
  const bf16* X;
  __device__ const bf16* rowptr(int m, int z) const {
    int b = m / kT, t = m - b * kT;
    return X + ((long)b * kL + 1 + (long)z * kT + t) * kH;
  }
};
struct AMapRows {           // plain strided rows
  const bf16* A; long stride;
  __device__ const bf16* rowptr(int m, int) const { return A + (long)m * stride; }
};

struct StQK {               // out[b][z][h][t][d] bf16
  bf16* out;
  __device__ void store(int m, int n, int z, float v) const {
    if (m >= kB * kT) return;
    int b = m / kT, t = m - b * kT, h = n >> 6, d = n & 63;
    out[((((long)b * kSEG + z) * kNH + h) * kT + t) * kDH + d] = (bf16)v;
  }
};
struct StVt {               // transposed V: out[b][z][h][d][t] bf16, t-stride kTP
  bf16* out;
  __device__ void store(int m, int n, int z, float v) const {
    if (m >= kB * kT) return;
    int b = m / kT, t = m - b * kT, h = n >> 6, d = n & 63;
    out[((((long)b * kSEG + z) * kNH + h) * kDH + d) * kTP + t] = (bf16)v;
  }
};
struct StRowF {             // plain fp32 row-major [M][1024], optional bias
  float* out; const float* bias; int M;
  __device__ void store(int m, int n, int, float v) const {
    if (m >= M) return;
    out[(long)m * kH + n] = v + (bias ? bias[n] : 0.0f);
  }
};
struct StOutProj {          // h = acc + out_b[n] + residual  (pre-LayerNorm)
  float* out; const float* bias; const float* resid; int M;
  __device__ void store(int m, int n, int, float v) const {
    if (m >= M) return;
    long idx = (long)m * kH + n;
    out[idx] = v + bias[n] + resid[idx];
  }
};

template<class AMap, class CSt>
__global__ __launch_bounds__(256)
void gemm_bf16_k1024(AMap amap, const bf16* __restrict__ W, long wstride, CSt cst, int M) {
  __shared__ __align__(16) bf16 As[128 * 32];
  __shared__ __align__(16) bf16 Bs[128 * 32];

  const int tid   = threadIdx.x;
  const int lane  = tid & 31;
  const int wid   = tid >> 5;        // 0..7
  const int waveM = wid >> 2;        // 0..1
  const int waveN = wid & 3;         // 0..3
  const int z  = blockIdx.z;
  const int m0 = blockIdx.x * 128;
  const int n0 = blockIdx.y * 128;

  v8f acc[4][2];
#pragma unroll
  for (int mt = 0; mt < 4; ++mt)
#pragma unroll
    for (int nt = 0; nt < 2; ++nt)
#pragma unroll
      for (int i = 0; i < 8; ++i) acc[mt][nt][i] = 0.0f;

  const int ldRow  = tid >> 1;          // 0..127
  const int ldHalf = (tid & 1) * 16;    // 0 or 16 elements
  int am = m0 + ldRow; if (am > M - 1) am = M - 1;
  const bf16* arow = amap.rowptr(am, z);
  const bf16* brow = W + (long)z * wstride + (long)(n0 + ldRow) * kH;

  const int koffA = (lane >> 4) * 8;
  const int koffB = (lane >> 4) * 16;

  // --- register double buffer: preload K-slice 0 ---
  v8bf ra0 = *(const v8bf*)(arow + ldHalf);
  v8bf ra1 = *(const v8bf*)(arow + ldHalf + 8);
  v8bf rb0 = *(const v8bf*)(brow + ldHalf);
  v8bf rb1 = *(const v8bf*)(brow + ldHalf + 8);

  for (int k0 = 0; k0 < kH; k0 += 32) {
    __syncthreads();
    *(v8bf*)&As[ldRow * 32 + ldHalf]     = ra0;
    *(v8bf*)&As[ldRow * 32 + ldHalf + 8] = ra1;
    *(v8bf*)&Bs[ldRow * 32 + ldHalf]     = rb0;
    *(v8bf*)&Bs[ldRow * 32 + ldHalf + 8] = rb1;
    __syncthreads();

    // Issue next slice's global loads now; latency hides under the WMMAs.
    if (k0 + 32 < kH) {
      ra0 = *(const v8bf*)(arow + k0 + 32 + ldHalf);
      ra1 = *(const v8bf*)(arow + k0 + 32 + ldHalf + 8);
      rb0 = *(const v8bf*)(brow + k0 + 32 + ldHalf);
      rb1 = *(const v8bf*)(brow + k0 + 32 + ldHalf + 8);
    }
    if (k0 + 64 < kH) {                       // global_prefetch_b8 two slices ahead
      __builtin_prefetch(arow + k0 + 64 + ldHalf, 0, 0);
      __builtin_prefetch(brow + k0 + 64 + ldHalf, 0, 0);
    }

    V16U a[4], b[2];
#pragma unroll
    for (int mt = 0; mt < 4; ++mt) {
      int r = waveM * 64 + mt * 16 + (lane & 15);
      a[mt].h[0] = *(const v8bf*)&As[r * 32 + koffA];
      a[mt].h[1] = *(const v8bf*)&As[r * 32 + 16 + koffA];
    }
#pragma unroll
    for (int nt = 0; nt < 2; ++nt) {
      int r = waveN * 32 + nt * 16 + (lane & 15);
      b[nt].h[0] = *(const v8bf*)&Bs[r * 32 + koffA];
      b[nt].h[1] = *(const v8bf*)&Bs[r * 32 + 16 + koffA];
    }
#pragma unroll
    for (int mt = 0; mt < 4; ++mt)
#pragma unroll
      for (int nt = 0; nt < 2; ++nt)
        acc[mt][nt] = wmma_bf16(a[mt], b[nt], acc[mt][nt]);
  }

#pragma unroll
  for (int mt = 0; mt < 4; ++mt)
#pragma unroll
    for (int nt = 0; nt < 2; ++nt)
#pragma unroll
      for (int i = 0; i < 8; ++i) {
        int m = m0 + waveM * 64 + mt * 16 + (lane >> 4) * 8 + i;
        int n = n0 + waveN * 32 + nt * 16 + (lane & 15);
        cst.store(m, n, z, acc[mt][nt][i]);
      }
}

// ---------------------------------------------------------------------------
// Fused per-segment attention.  One wave per (q-tile, head, b*seg).
// Phase 1: S = scale * Q K^T (WMMA, per 16x16 tile over 17 k-tiles)
// Phase 2: exact softmax over 257 keys (wave32 shuffle reductions)
// Phase 3: O = P V (WMMA, K-dim padded to 288, P zeroed beyond 257)
// ---------------------------------------------------------------------------
__global__ __launch_bounds__(32)
void seg_attn(const bf16* __restrict__ Q, const bf16* __restrict__ K,
              const bf16* __restrict__ Vt, float* __restrict__ self_out) {
  __shared__ __align__(16) float Sld[16][kTP];
  __shared__ __align__(16) bf16  Pld[16][kTP];

  const int lane = threadIdx.x;
  const int qt = blockIdx.x;            // 0..16
  const int h  = blockIdx.y;            // 0..15
  const int bs = blockIdx.z;            // 0..127
  const int s = bs & 15, b = bs >> 4;

  const long slab = ((long)b * kSEG + s) * kNH + h;
  const bf16* Qs = Q  + slab * kT * kDH;
  const bf16* Ks = K  + slab * kT * kDH;
  const bf16* Vs = Vt + slab * kDH * kTP;

  const int koffA = (lane >> 4) * 8;
  const int koffB = (lane >> 4) * 16;
  int qr = qt * 16 + (lane & 15); if (qr > kT - 1) qr = kT - 1;

  // ----- Phase 1: scores into LDS -----
  for (int kt = 0; kt < 17; ++kt) {
    v8f sa;
#pragma unroll
    for (int i = 0; i < 8; ++i) sa[i] = 0.0f;
#pragma unroll
    for (int kk = 0; kk < 2; ++kk) {
      V16U a, bb;
      a.h[0] = *(const v8bf*)(Qs + (long)qr * kDH + kk * 32 + koffA);
      a.h[1] = *(const v8bf*)(Qs + (long)qr * kDH + kk * 32 + 16 + koffA);
      int kc = kt * 16 + (lane & 15); if (kc > kT - 1) kc = kT - 1;
      bb.h[0] = *(const v8bf*)(Ks + (long)kc * kDH + kk * 32 + koffA);
      bb.h[1] = *(const v8bf*)(Ks + (long)kc * kDH + kk * 32 + 16 + koffA);
      sa = wmma_bf16(a, bb, sa);
    }
#pragma unroll
    for (int i = 0; i < 8; ++i) {
      int r = (lane >> 4) * 8 + i;
      int c = kt * 16 + (lane & 15);
      Sld[r][c] = sa[i] * 0.125f;   // 1/sqrt(64)
    }
  }
  __syncthreads();

  // ----- Phase 2: softmax (exact, 257 valid keys), P in bf16 -----
  for (int r = 0; r < 16; ++r) {
    float mx = -3.0e38f;
    for (int c = lane; c < kT; c += 32) mx = fmaxf(mx, Sld[r][c]);
#pragma unroll
    for (int off = 16; off; off >>= 1) mx = fmaxf(mx, __shfl_xor(mx, off, 32));
    float sum = 0.0f;
    for (int c = lane; c < kT; c += 32) sum += __expf(Sld[r][c] - mx);
#pragma unroll
    for (int off = 16; off; off >>= 1) sum += __shfl_xor(sum, off, 32);
    float inv = 1.0f / sum;
    for (int c = lane; c < kTP; c += 32)
      Pld[r][c] = (c < kT) ? (bf16)(__expf(Sld[r][c] - mx) * inv) : (bf16)0.0f;
  }
  __syncthreads();

  // ----- Phase 3: O = P @ V (Vt is [d][t], t padded to 288) -----
#pragma unroll
  for (int nt = 0; nt < 4; ++nt) {
    v8f oa;
#pragma unroll
    for (int i = 0; i < 8; ++i) oa[i] = 0.0f;
    for (int ks = 0; ks < 9; ++ks) {
      V16U a, bb;
      int pr = lane & 15;
      a.h[0] = *(const v8bf*)&Pld[pr][ks * 32 + koffA];
      a.h[1] = *(const v8bf*)&Pld[pr][ks * 32 + 16 + koffA];
      int d = nt * 16 + (lane & 15);
      bb.h[0] = *(const v8bf*)(Vs + (long)d * kTP + ks * 32 + koffA);
      bb.h[1] = *(const v8bf*)(Vs + (long)d * kTP + ks * 32 + 16 + koffA);
      oa = wmma_bf16(a, bb, oa);
    }
#pragma unroll
    for (int i = 0; i < 8; ++i) {
      int q = qt * 16 + (lane >> 4) * 8 + i;
      if (q < kT) {
        int d = nt * 16 + (lane & 15);
        self_out[((long)b * kL + 1 + (long)s * kT + q) * kH + h * kDH + d] = oa[i];
      }
    }
  }
}

// ---------------------------------------------------------------------------
// Lord path (tiny): gather -> (GEMMs above) -> 17-token attention -> scatter
// ---------------------------------------------------------------------------
__global__ void k_gather_lords(const float* __restrict__ blord,
                               const float* __restrict__ self_out,
                               bf16* __restrict__ lords_in) {
  long i = (long)blockIdx.x * blockDim.x + threadIdx.x;
  long n = (long)kB * 17 * kH;
  if (i >= n) return;
  int c = (int)(i & (kH - 1));
  int r = (int)(i >> 10);
  int b = r / 17, t = r % 17;
  float v = (t == 0) ? blord[(long)b * kH + c]
                     : self_out[((long)b * kL + 1 + (long)(t - 1) * kT) * kH + c];
  lords_in[i] = (bf16)v;
}

__global__ __launch_bounds__(32)
void lord_attn(const float* __restrict__ lq, const float* __restrict__ lk,
               const float* __restrict__ lv, float* __restrict__ lout) {
  int b = blockIdx.x >> 4, h = blockIdx.x & 15;
  int i = threadIdx.x;
  if (i >= 17) return;
  const float* qrow = lq + ((long)(b * 17 + i) * kH) + h * kDH;
  float sc[17], mx = -3.0e38f;
  for (int j = 0; j < 17; ++j) {
    const float* krow = lk + ((long)(b * 17 + j) * kH) + h * kDH;
    float sv = 0.0f;
    for (int d = 0; d < kDH; ++d) sv += qrow[d] * krow[d];
    sc[j] = sv * 0.125f;
    mx = fmaxf(mx, sc[j]);
  }
  float sum = 0.0f;
  for (int j = 0; j < 17; ++j) { sc[j] = __expf(sc[j] - mx); sum += sc[j]; }
  float inv = 1.0f / sum;
  for (int d = 0; d < kDH; ++d) {
    float o = 0.0f;
    for (int j = 0; j < 17; ++j)
      o += sc[j] * lv[((long)(b * 17 + j) * kH) + h * kDH + d];
    lout[((long)(b * 17 + i) * kH) + h * kDH + d] = o * inv;
  }
}

__global__ void k_scatter_lords(const float* __restrict__ lout, float* __restrict__ self_out) {
  long i = (long)blockIdx.x * blockDim.x + threadIdx.x;
  long n = (long)kB * 17 * kH;
  if (i >= n) return;
  int c = (int)(i & (kH - 1));
  int r = (int)(i >> 10);
  int b = r / 17, t = r % 17;
  long drow = (t == 0) ? (long)b * kL : (long)b * kL + 1 + (long)(t - 1) * kT;
  self_out[drow * kH + c] = lout[i];
}

// ---------------------------------------------------------------------------
// In-place LayerNorm over rows of 1024 (one block per row)
// ---------------------------------------------------------------------------
__global__ __launch_bounds__(256)
void layernorm_inplace(float* __restrict__ hbuf, const float* __restrict__ w,
                       const float* __restrict__ bsh) {
  __shared__ float red[256];
  long row = blockIdx.x;
  float* p = hbuf + row * kH;
  int tid = threadIdx.x;
  float x[4];
#pragma unroll
  for (int j = 0; j < 4; ++j) x[j] = p[tid + j * 256];
  float s = x[0] + x[1] + x[2] + x[3];
  red[tid] = s; __syncthreads();
  for (int o = 128; o; o >>= 1) { if (tid < o) red[tid] += red[tid + o]; __syncthreads(); }
  float mu = red[0] * (1.0f / kH);
  __syncthreads();
  float sq = 0.0f;
#pragma unroll
  for (int j = 0; j < 4; ++j) { float d = x[j] - mu; sq += d * d; }
  red[tid] = sq; __syncthreads();
  for (int o = 128; o; o >>= 1) { if (tid < o) red[tid] += red[tid + o]; __syncthreads(); }
  float var = red[0] * (1.0f / kH);
  float rstd = rsqrtf(var + LN_EPS);
#pragma unroll
  for (int j = 0; j < 4; ++j) {
    int c = tid + j * 256;
    p[c] = (x[j] - mu) * rstd * w[c] + bsh[c];
  }
}

// ---------------------------------------------------------------------------
// Host: workspace carving + launch sequence (all on `stream`, graph-safe)
// ---------------------------------------------------------------------------
extern "C" void kernel_launch(void* const* d_in, const int* in_sizes, int n_in,
                              void* d_out, int out_size, void* d_ws, size_t ws_size,
                              hipStream_t stream) {
  (void)in_sizes; (void)n_in; (void)out_size; (void)ws_size;
  const float* hidden  = (const float*)d_in[0];
  const float* Blord_w = (const float*)d_in[1];
  const float* Blord_b = (const float*)d_in[2];
  const float* seg_qw  = (const float*)d_in[3];
  const float* seg_kw  = (const float*)d_in[4];
  const float* seg_vw  = (const float*)d_in[5];
  const float* lord_qw = (const float*)d_in[6];
  const float* lord_kw = (const float*)d_in[7];
  const float* lord_vw = (const float*)d_in[8];
  const float* out_w   = (const float*)d_in[9];
  const float* out_b   = (const float*)d_in[10];
  const float* ln_w    = (const float*)d_in[11];
  const float* ln_b    = (const float*)d_in[12];

  const long NBLH = (long)kB * kL * kH;              // 33,693,696
  const long SEGHH = (long)kSEG * kH * kH;           // 16,777,216
  const long HH = (long)kH * kH;
  const long QN = (long)kB * kSEG * kNH * kT * kDH;  // 33,685,504
  const long VtN = (long)kB * kSEG * kNH * kDH * kTP;// 37,748,736
  const long LORDN = (long)kB * 17 * kH;             // 139,264

  char* w = (char*)d_ws;
  auto carve = [&](size_t bytes) { char* p = w; w += (bytes + 255) & ~(size_t)255; return p; };
  bf16* Xb      = (bf16*)carve(NBLH * 2);
  bf16* Wqb     = (bf16*)carve(SEGHH * 2);
  bf16* Wkb     = (bf16*)carve(SEGHH * 2);
  bf16* Wvb     = (bf16*)carve(SEGHH * 2);
  bf16* Wblordb = (bf16*)carve(HH * 2);
  bf16* Wlqb    = (bf16*)carve(HH * 2);
  bf16* Wlkb    = (bf16*)carve(HH * 2);
  bf16* Wlvb    = (bf16*)carve(HH * 2);
  bf16* Woutb   = (bf16*)carve(HH * 2);
  bf16* Qb      = (bf16*)carve(QN * 2);
  bf16* Kb      = (bf16*)carve(QN * 2);
  bf16* Vtb     = (bf16*)carve(VtN * 2);
  float* Blordf = (float*)carve((size_t)kB * kH * 4);
  bf16* LordsInB = (bf16*)carve(LORDN * 2);
  float* lqf    = (float*)carve(LORDN * 4);
  float* lkf    = (float*)carve(LORDN * 4);
  float* lvf    = (float*)carve(LORDN * 4);
  float* loutf  = (float*)carve(LORDN * 4);
  bf16* Sb      = (bf16*)carve(NBLH * 2);

  float* att_out  = (float*)d_out;          // [B,L,H] attention_output (built in place)
  float* self_out = (float*)d_out + NBLH;   // [B,L,H] self_outputs

  const int CB = 256, CG = 4096;
  // --- bf16 conversions of activations + all weights ---
  k_f32_to_bf16<<<CG, CB, 0, stream>>>(hidden, Xb, NBLH);
  k_f32_to_bf16<<<CG, CB, 0, stream>>>(seg_qw, Wqb, SEGHH);
  k_f32_to_bf16<<<CG, CB, 0, stream>>>(seg_kw, Wkb, SEGHH);
  k_f32_to_bf16<<<CG, CB, 0, stream>>>(seg_vw, Wvb, SEGHH);
  k_f32_to_bf16<<<CG, CB, 0, stream>>>(Blord_w, Wblordb, HH);
  k_f32_to_bf16<<<CG, CB, 0, stream>>>(lord_qw, Wlqb, HH);
  k_f32_to_bf16<<<CG, CB, 0, stream>>>(lord_kw, Wlkb, HH);
  k_f32_to_bf16<<<CG, CB, 0, stream>>>(lord_vw, Wlvb, HH);
  k_f32_to_bf16<<<CG, CB, 0, stream>>>(out_w, Woutb, HH);
  k_fill_bf16_zero<<<CG, CB, 0, stream>>>(Vtb, VtN);   // zero t-padding for P@V

  // --- per-segment QKV projections (big WMMA GEMMs), M = B*T = 2056 ---
  dim3 gseg((kB * kT + 127) / 128, kH / 128, kSEG);
  gemm_bf16_k1024<AMapSegX, StQK><<<gseg, 256, 0, stream>>>(
      AMapSegX{Xb}, Wqb, HH, StQK{Qb}, kB * kT);
  gemm_bf16_k1024<AMapSegX, StQK><<<gseg, 256, 0, stream>>>(
      AMapSegX{Xb}, Wkb, HH, StQK{Kb}, kB * kT);
  gemm_bf16_k1024<AMapSegX, StVt><<<gseg, 256, 0, stream>>>(
      AMapSegX{Xb}, Wvb, HH, StVt{Vtb}, kB * kT);

  // --- fused per-segment attention (writes seg_out into self_outputs) ---
  seg_attn<<<dim3(17, kNH, kB * kSEG), 32, 0, stream>>>(Qb, Kb, Vtb, self_out);

  // --- Blord projection of token 0: M = 8 ---
  gemm_bf16_k1024<AMapRows, StRowF><<<dim3(1, kH / 128, 1), 256, 0, stream>>>(
      AMapRows{Xb, (long)kL * kH}, Wblordb, 0, StRowF{Blordf, Blord_b, kB}, kB);

  // --- lord attention over 17 tokens per batch ---
  int lgrid = (int)((LORDN + CB - 1) / CB);
  k_gather_lords<<<lgrid, CB, 0, stream>>>(Blordf, self_out, LordsInB);
  dim3 glord((kB * 17 + 127) / 128, kH / 128, 1);
  gemm_bf16_k1024<AMapRows, StRowF><<<glord, 256, 0, stream>>>(
      AMapRows{LordsInB, kH}, Wlqb, 0, StRowF{lqf, nullptr, kB * 17}, kB * 17);
  gemm_bf16_k1024<AMapRows, StRowF><<<glord, 256, 0, stream>>>(
      AMapRows{LordsInB, kH}, Wlkb, 0, StRowF{lkf, nullptr, kB * 17}, kB * 17);
  gemm_bf16_k1024<AMapRows, StRowF><<<glord, 256, 0, stream>>>(
      AMapRows{LordsInB, kH}, Wlvb, 0, StRowF{lvf, nullptr, kB * 17}, kB * 17);
  lord_attn<<<kB * kNH, 32, 0, stream>>>(lqf, lkf, lvf, loutf);
  k_scatter_lords<<<lgrid, CB, 0, stream>>>(loutf, self_out);

  // --- output dense + bias + residual (pre-LN into d_out[0:NBLH]) ---
  k_f32_to_bf16<<<CG, CB, 0, stream>>>(self_out, Sb, NBLH);
  dim3 gout((kB * kL + 127) / 128, kH / 128, 1);
  gemm_bf16_k1024<AMapRows, StOutProj><<<gout, 256, 0, stream>>>(
      AMapRows{Sb, kH}, Woutb, 0, StOutProj{att_out, out_b, hidden, kB * kL}, kB * kL);

  // --- LayerNorm in place on d_out first half ---
  layernorm_inplace<<<kB * kL, 256, 0, stream>>>(att_out, ln_w, ln_b);
}
